// LongformerAttention_5076651344512
// MI455X (gfx1250) — matrix operand order repaired and checked by reading, not compiled, not run on hardware
//
#include <hip/hip_runtime.h>

typedef __bf16 bf16_t;
typedef __attribute__((ext_vector_type(16))) __bf16 v16bf;
typedef __attribute__((ext_vector_type(8)))  __bf16 v8bf;
typedef __attribute__((ext_vector_type(8)))  float  v8f;

#define D_MODEL   1024
#define NUM_HEADS 16
#define D_HEAD    64
#define SEQ       2048
#define BATCH     2
#define WIN_HALF  256
#define NGLOBAL   1
#define MTOT      (BATCH * SEQ)   // 4096 rows
#define LOG2E     1.44269504088896340f

// ---------------------------------------------------------------------------
// WMMA helper:  D = A(16x32 bf16) * B(32x16 bf16) + C(16x16 f32)
// ---------------------------------------------------------------------------
__device__ __forceinline__ v8f wmma_bf16(v16bf a, v16bf b, v8f c) {
  return __builtin_amdgcn_wmma_f32_16x16x32_bf16(
      /*neg_a=*/false, a, /*neg_b=*/false, b,
      /*c_mod=*/(short)0, c, /*reuse_a=*/false, /*reuse_b=*/false);
}

__device__ __forceinline__ v16bf cat16(v8bf lo, v8bf hi) {
  return __builtin_shufflevector(lo, hi, 0,1,2,3,4,5,6,7,8,9,10,11,12,13,14,15);
}

// A-fragment (16x32 bf16): lane L -> row L%16; K halves {kb..kb+7, kb+16..kb+23},
// kb = 8*(L>=16).  `row` points at the 32-element K-chunk start of this lane's row.
__device__ __forceinline__ v16bf load_a_frag(const bf16_t* __restrict__ row, int kb) {
  v8bf lo = *(const v8bf*)(row + kb);
  v8bf hi = *(const v8bf*)(row + kb + 16);
  return cat16(lo, hi);
}

// ---------------------------------------------------------------------------
// f32 -> bf16 elementwise convert
// ---------------------------------------------------------------------------
__global__ __launch_bounds__(256) void k_cvt_bf16(const float* __restrict__ in,
                                                  bf16_t* __restrict__ out, int n) {
  int i = blockIdx.x * 256 + threadIdx.x;
  if (i < n) out[i] = (bf16_t)in[i];
}

// ---------------------------------------------------------------------------
// 1024x1024 transpose + convert: Wt[o][i] = (bf16) W[i][o]   (LDS 32x32 tiles)
// ---------------------------------------------------------------------------
__global__ __launch_bounds__(256) void k_transpose_w(const float* __restrict__ W,
                                                     bf16_t* __restrict__ Wt) {
  __shared__ bf16_t tile[32][33];
  int bx = blockIdx.x & 31;          // column-block of W
  int by = blockIdx.x >> 5;          // row-block of W
  int tx = threadIdx.x & 31;
  int ty = threadIdx.x >> 5;         // 0..7
#pragma unroll
  for (int yy = ty; yy < 32; yy += 8)
    tile[yy][tx] = (bf16_t)W[(by * 32 + yy) * D_MODEL + bx * 32 + tx];
  __syncthreads();
#pragma unroll
  for (int yy = ty; yy < 32; yy += 8)
    Wt[(bx * 32 + yy) * D_MODEL + by * 32 + tx] = tile[tx][yy];
}

// ---------------------------------------------------------------------------
// bf16 WMMA GEMM:  C(M=4096 x N=1024) = A(4096x1024) * Bt^T
//   A    row-major [m][k]
//   Bt   row-major [n][k]   (pre-transposed weights -> contiguous K per lane)
// One wave computes a 32x64 tile (2 M-subtiles x 4 N-subtiles = 8 C fragments),
// so each B fragment is reused by two WMMAs: 6 fragment loads -> 8 WMMAs per k-step.
// 4 waves / block; 2048 wave tiles -> 512 blocks.
// mode: 0 = Q (scaled, [b,h,s,d] bf16)   1 = K ([b,h,s,d] bf16)
//       2 = V ([b,h,d,s] bf16, transposed)  3 = f32 out + bias
// ---------------------------------------------------------------------------
__global__ __launch_bounds__(128)
void k_gemm_bf16(const bf16_t* __restrict__ A, const bf16_t* __restrict__ Bt,
                 void* __restrict__ Out, const float* __restrict__ bias,
                 int mode, float scale) {
  const int lane  = threadIdx.x & 31;
  const int wave  = threadIdx.x >> 5;
  const int tile  = blockIdx.x * 4 + wave;      // 2048 tiles
  const int mt    = tile & 127;                 // 128 M tiles of 32 (fastest: B reuse)
  const int nt    = tile >> 7;                  // 16 N tiles of 64
  const int mbase = mt * 32;
  const int nbase = nt * 64;
  const int lmod  = lane & 15;
  const int kb    = (lane < 16) ? 0 : 8;
  const int khalf = (lane < 16) ? 0 : 16;
  const int rowadd = (lane < 16) ? 0 : 8;

  v8f acc[2][4];
#pragma unroll
  for (int mi = 0; mi < 2; ++mi)
#pragma unroll
    for (int dt = 0; dt < 4; ++dt)
      acc[mi][dt] = (v8f){0.f,0.f,0.f,0.f,0.f,0.f,0.f,0.f};

  const bf16_t* arow0 = A + (size_t)(mbase + lmod) * D_MODEL;
  const bf16_t* arow1 = arow0 + (size_t)16 * D_MODEL;
  for (int k0 = 0; k0 < D_MODEL; k0 += 32) {
    v16bf af0 = load_a_frag(arow0 + k0, kb);
    v16bf af1 = load_a_frag(arow1 + k0, kb);
#pragma unroll
    for (int dt = 0; dt < 4; ++dt) {
      v16bf bfr = load_a_frag(Bt + (size_t)(nbase + dt * 16 + lmod) * D_MODEL + k0, kb);
      acc[0][dt] = wmma_bf16(af0, bfr, acc[0][dt]);
      acc[1][dt] = wmma_bf16(af1, bfr, acc[1][dt]);
    }
  }

#pragma unroll
  for (int mi = 0; mi < 2; ++mi) {
#pragma unroll
    for (int dt = 0; dt < 4; ++dt) {
#pragma unroll
      for (int r = 0; r < 8; ++r) {
        int grow = mbase + mi * 16 + r + rowadd;  // global row (b*S + s)
        int gcol = nbase + dt * 16 + lmod;        // global col (h*64 + d)
        float v = acc[mi][dt][r] * scale;
        if (mode == 3) {
          ((float*)Out)[(size_t)grow * D_MODEL + gcol] = v + bias[gcol];
        } else {
          int b = grow >> 11, s = grow & (SEQ - 1);
          int h = gcol >> 6,  d = gcol & (D_HEAD - 1);
          bf16_t bv = (bf16_t)v;
          if (mode == 2)
            ((bf16_t*)Out)[((size_t)(b * NUM_HEADS + h) * D_HEAD + d) * SEQ + s] = bv;
          else
            ((bf16_t*)Out)[((size_t)(b * NUM_HEADS + h) * SEQ + s) * D_HEAD + d] = bv;
        }
      }
    }
  }
}

// ---------------------------------------------------------------------------
// Banded (Longformer) flash attention.
// Grid: 1024 WGs = B(2) * H(16) * 32 query-tiles of 64.  4 waves x 16 queries.
// Q is pre-scaled by (1/8)*log2(e): softmax runs in exp2 domain.
// ---------------------------------------------------------------------------
__global__ __launch_bounds__(256)
void k_attn(const bf16_t* __restrict__ Qb, const bf16_t* __restrict__ Kb,
            const bf16_t* __restrict__ Vt, bf16_t* __restrict__ AttnOut) {
  __shared__ __align__(32) bf16_t Plds[4][16 * 32];

  const int lane   = threadIdx.x & 31;
  const int wave   = threadIdx.x >> 5;
  const int wg     = blockIdx.x;
  const int qt     = wg & 31;
  const int h      = (wg >> 5) & 15;
  const int b      = wg >> 9;
  const int qbwg   = qt * 64;            // workgroup query base
  const int qbase  = qbwg + wave * 16;   // this wave's query base
  const int lmod   = lane & 15;
  const int kb8    = (lane < 16) ? 0 : 8;
  const int khalf  = (lane < 16) ? 0 : 16;
  const int rowadd = kb8;                // C-fragment row offset

  const bf16_t* Qh = Qb + (size_t)(b * NUM_HEADS + h) * SEQ * D_HEAD;
  const bf16_t* Kh = Kb + (size_t)(b * NUM_HEADS + h) * SEQ * D_HEAD;
  const bf16_t* Vh = Vt + (size_t)(b * NUM_HEADS + h) * D_HEAD * SEQ;

  // Preload Q A-fragments (16 queries x 64 d  ->  two 16x32 fragments).
  const bf16_t* qrow = Qh + (size_t)(qbase + lmod) * D_HEAD;
  v16bf qa0 = load_a_frag(qrow,      kb8);
  v16bf qa1 = load_a_frag(qrow + 32, kb8);

  float mrun[8], lrun[8];
  v8f acc[4];
#pragma unroll
  for (int r = 0; r < 8; ++r) { mrun[r] = -1e30f; lrun[r] = 0.f; }
#pragma unroll
  for (int dt = 0; dt < 4; ++dt) acc[dt] = (v8f){0.f,0.f,0.f,0.f,0.f,0.f,0.f,0.f};

  // Key sweep bounds (uniform per workgroup).
  int jlo, jhi;
  if (qbwg < NGLOBAL) { jlo = 0; jhi = SEQ; }               // tile holds global row 0
  else {
    jlo = qbwg - WIN_HALF; if (jlo < 0) jlo = 0; jlo &= ~31;
    jhi = qbwg + 64 - 1 + WIN_HALF + 1; if (jhi > SEQ) jhi = SEQ;
    jhi = (jhi + 31) & ~31;
  }
  const bool extra0 = (jlo > 0);   // extra block for global column j=0 (processed first)
  const int  nsteps = (jhi - jlo) / 32 + (extra0 ? 1 : 0);

  bf16_t* pl = Plds[wave];

  for (int step = 0; step < nsteps; ++step) {
    int j0 = (extra0 && step == 0) ? 0 : jlo + (step - (extra0 ? 1 : 0)) * 32;

    // ---- prefetch next key block (global_prefetch_b8: no counter, pure latency hide)
    if (step + 1 < nsteps) {
      int jn = (extra0 && step == 0) ? jlo : j0 + 32;
      __builtin_prefetch(Kh + (size_t)(jn + lmod) * D_HEAD, 0, 3);
      __builtin_prefetch(Kh + (size_t)(jn + 16 + lmod) * D_HEAD, 0, 3);
#pragma unroll
      for (int dt = 0; dt < 4; ++dt)
        __builtin_prefetch(Vh + (size_t)(dt * 16 + lmod) * SEQ + jn, 0, 3);
    }

    // ---- scores: S(16x32) = Q(16x64) * K^T(64x32)  -> two 16x16 fragments
    v8f sf[2];
#pragma unroll
    for (int half = 0; half < 2; ++half) {
      const bf16_t* krow = Kh + (size_t)(j0 + half * 16 + lmod) * D_HEAD;
      v16bf kf0 = load_a_frag(krow, kb8);
      v16bf kf1 = load_a_frag(krow + 32, kb8);
      v8f s = (v8f){0.f,0.f,0.f,0.f,0.f,0.f,0.f,0.f};
      s = wmma_bf16(qa0, kf0, s);
      s = wmma_bf16(qa1, kf1, s);
      sf[half] = s;
    }

    // ---- Longformer mask
#pragma unroll
    for (int half = 0; half < 2; ++half) {
      int j = j0 + half * 16 + lmod;
#pragma unroll
      for (int r = 0; r < 8; ++r) {
        int q = qbase + r + rowadd;
        int d = q - j; if (d < 0) d = -d;
        bool ok = (d <= WIN_HALF) || (q < NGLOBAL) || (j < NGLOBAL);
        if (!ok) sf[half][r] = -1e30f;
      }
    }

    // ---- online softmax (row reductions across the 16-lane half-wave)
    float cfac[8];
#pragma unroll
    for (int r = 0; r < 8; ++r) {
      float mx = fmaxf(sf[0][r], sf[1][r]);
#pragma unroll
      for (int off = 8; off >= 1; off >>= 1) mx = fmaxf(mx, __shfl_xor(mx, off, 32));
      float mnew = fmaxf(mrun[r], mx);
      float c  = exp2f(mrun[r] - mnew);
      float p0 = exp2f(sf[0][r] - mnew);
      float p1 = exp2f(sf[1][r] - mnew);
      sf[0][r] = p0; sf[1][r] = p1;
      float rs = p0 + p1;
#pragma unroll
      for (int off = 8; off >= 1; off >>= 1) rs += __shfl_xor(rs, off, 32);
      lrun[r] = lrun[r] * c + rs;
      mrun[r] = mnew;
      cfac[r] = c;
    }

    // ---- C-layout -> A-layout via LDS (same-wave, in-order DS + dscnt wait)
#pragma unroll
    for (int half = 0; half < 2; ++half) {
      int n = half * 16 + lmod;
#pragma unroll
      for (int r = 0; r < 8; ++r)
        pl[(r + rowadd) * 32 + n] = (bf16_t)sf[half][r];
    }
    asm volatile("s_wait_dscnt 0" ::: "memory");
    v16bf pa = load_a_frag(pl + lmod * 32, kb8);

    // ---- O = O*c + P(16x32) * V(32x64)   (V stored [d][s]: contiguous keys)
#pragma unroll
    for (int dt = 0; dt < 4; ++dt) {
      v16bf vf = load_a_frag(Vh + (size_t)(dt * 16 + lmod) * SEQ + j0, kb8);
#pragma unroll
      for (int r = 0; r < 8; ++r) acc[dt][r] *= cfac[r];
      acc[dt] = wmma_bf16(pa, vf, acc[dt]);
    }
  }

  // ---- normalize and store as bf16 [b][s][h*64+d] for the output projection
  float inv[8];
#pragma unroll
  for (int r = 0; r < 8; ++r) inv[r] = 1.0f / lrun[r];
#pragma unroll
  for (int dt = 0; dt < 4; ++dt) {
#pragma unroll
    for (int r = 0; r < 8; ++r) {
      int q = qbase + r + rowadd;
      int gcol = h * D_HEAD + dt * 16 + lmod;
      AttnOut[(size_t)(b * SEQ + q) * D_MODEL + gcol] = (bf16_t)(acc[dt][r] * inv[r]);
    }
  }
}

// ---------------------------------------------------------------------------
// Host-side orchestration
// ---------------------------------------------------------------------------
extern "C" void kernel_launch(void* const* d_in, const int* in_sizes, int n_in,
                              void* d_out, int out_size, void* d_ws, size_t ws_size,
                              hipStream_t stream) {
  (void)in_sizes; (void)n_in; (void)out_size; (void)ws_size;
  const float* hidden = (const float*)d_in[0];
  const float* Wq = (const float*)d_in[1];
  const float* Wk = (const float*)d_in[2];
  const float* Wv = (const float*)d_in[3];
  const float* Wo = (const float*)d_in[4];
  const float* bo = (const float*)d_in[5];

  // Workspace layout (bf16 elements), total 48 MB.
  bf16_t* ws  = (bf16_t*)d_ws;
  size_t  off = 0;
  bf16_t* Xbf  = ws + off; off += (size_t)MTOT * D_MODEL;        // 4M
  bf16_t* Wqt  = ws + off; off += (size_t)D_MODEL * D_MODEL;     // 1M
  bf16_t* Wkt  = ws + off; off += (size_t)D_MODEL * D_MODEL;
  bf16_t* Wvt  = ws + off; off += (size_t)D_MODEL * D_MODEL;
  bf16_t* Wot  = ws + off; off += (size_t)D_MODEL * D_MODEL;
  bf16_t* Qb   = ws + off; off += (size_t)MTOT * D_MODEL;        // [b,h,s,d]
  bf16_t* Kb   = ws + off; off += (size_t)MTOT * D_MODEL;        // [b,h,s,d]
  bf16_t* Vtb  = ws + off; off += (size_t)MTOT * D_MODEL;        // [b,h,d,s]
  bf16_t* Attn = ws + off; off += (size_t)MTOT * D_MODEL;        // [b,s,h*64+d]

  const int nX = MTOT * D_MODEL;
  k_cvt_bf16<<<(nX + 255) / 256, 256, 0, stream>>>(hidden, Xbf, nX);
  k_transpose_w<<<1024, 256, 0, stream>>>(Wq, Wqt);
  k_transpose_w<<<1024, 256, 0, stream>>>(Wk, Wkt);
  k_transpose_w<<<1024, 256, 0, stream>>>(Wv, Wvt);
  k_transpose_w<<<1024, 256, 0, stream>>>(Wo, Wot);

  const float qscale = 0.125f * LOG2E;   // 1/sqrt(64), folded into exp2 domain
  k_gemm_bf16<<<512, 128, 0, stream>>>(Xbf, Wqt, Qb,  nullptr, 0, qscale);
  k_gemm_bf16<<<512, 128, 0, stream>>>(Xbf, Wkt, Kb,  nullptr, 1, 1.0f);
  k_gemm_bf16<<<512, 128, 0, stream>>>(Xbf, Wvt, Vtb, nullptr, 2, 1.0f);

  k_attn<<<1024, 256, 0, stream>>>(Qb, Kb, Vtb, Attn);

  k_gemm_bf16<<<512, 128, 0, stream>>>(Attn, Wot, d_out, bo, 3, 1.0f);
}
